// SSLModel_87754771792394
// MI455X (gfx1250) — matrix operand order repaired and checked by reading, not compiled, not run on hardware
//
#include <hip/hip_runtime.h>
#include <hip/hip_bf16.h>

typedef __attribute__((ext_vector_type(16))) _Float16 v16h;
typedef __attribute__((ext_vector_type(8)))  float    v8f;

// ---------------------------------------------------------------------------
// WMMA fragment index helpers (CDNA5 16x16x32 f16, wave32; contiguous k order used by this code, not the ISA table order):
//  A (16x32, M across lanes): lane m=l&15, half=l>>4; VGPR v holds K pair at
//      kA(v) = 2v + 16*half
//  B (32x16, N across lanes): lane n=l&15, half=l>>4; 16 contiguous f16 at
//      k in [16*half, 16*half+16)   -> single 32B contiguous LDS read
//  C/D (16x16 f32): row = v + 8*half, col = l&15
// ---------------------------------------------------------------------------
__device__ __forceinline__ int kA_of(int v, int half) {
  return 2 * v + (half << 4);
}

#define HID 256
#define ADIM 133
#define BDIM 14

// ============================ GEMM 1: h0 ====================================
// h0[e] = relu( [V[src[e]]; E[e]] @ W_i^T ),  K = 147 padded to 160
__global__ __launch_bounds__(256) void gemm_h0_kernel(
    const float* __restrict__ V, const float* __restrict__ Eb,
    const float* __restrict__ Wi, const int* __restrict__ src,
    float* __restrict__ h0, int nEdges) {
  extern __shared__ _Float16 smem[];            // [256][160] f16 row-major
  const int KPAD = 160, KREAL = ADIM + BDIM;    // 147
  for (int idx = threadIdx.x; idx < HID * KPAD; idx += blockDim.x) {
    int n = idx / KPAD, k = idx - n * KPAD;
    smem[idx] = (k < KREAL) ? (_Float16)Wi[n * KREAL + k] : (_Float16)0.0f;
  }
  __syncthreads();

  const int lane = threadIdx.x & 31, wave = threadIdx.x >> 5;
  const int half = lane >> 4, nl = lane & 15;
  const int rowBase = blockIdx.x * 128 + wave * 16;
  int row = rowBase + nl;
  if (row >= nEdges) row = nEdges - 1;          // clamp loads; stores masked
  const int se = src[row];
  const float* vrow = V + (size_t)se * ADIM;
  const float* erow = Eb + (size_t)row * BDIM;

  v8f zero = {0.f, 0.f, 0.f, 0.f, 0.f, 0.f, 0.f, 0.f};
  v8f acc[16];
#pragma unroll
  for (int nt = 0; nt < 16; ++nt) acc[nt] = zero;

  for (int kt = 0; kt < KPAD / 32; ++kt) {
    v16h a;
#pragma unroll
    for (int v = 0; v < 8; ++v) {
      int k = kt * 32 + kA_of(v, half);
      float x0 = (k < ADIM) ? vrow[k] : ((k < KREAL) ? erow[k - ADIM] : 0.f);
      int k1 = k + 1;
      float x1 = (k1 < ADIM) ? vrow[k1] : ((k1 < KREAL) ? erow[k1 - ADIM] : 0.f);
      a[2 * v]     = (_Float16)x0;
      a[2 * v + 1] = (_Float16)x1;
    }
#pragma unroll
    for (int nt = 0; nt < 16; ++nt) {
      const int ncol = nt * 16 + nl;
      v16h b = *(const v16h*)&smem[ncol * KPAD + kt * 32 + half * 16];
      acc[nt] = __builtin_amdgcn_wmma_f32_16x16x32_f16(
          false, a, false, b, (short)0, acc[nt], false, false);
    }
  }
#pragma unroll
  for (int nt = 0; nt < 16; ++nt) {
#pragma unroll
    for (int v = 0; v < 8; ++v) {
      int r = rowBase + v + 8 * half;
      if (r < nEdges) {
        float val = acc[nt][v];
        h0[(size_t)r * HID + nt * 16 + nl] = val > 0.f ? val : 0.f;
      }
    }
  }
}

// ========================= GEMM 2: message update ===========================
// m[e] = m_atom[src[e]] - w[rev[e]]*h[rev[e]];  hNext = relu(m @ W_h^T + h0)
__global__ __launch_bounds__(256) void gemm_update_kernel(
    const float* __restrict__ mAtom, const float* __restrict__ hCur,
    const float* __restrict__ h0, const float* __restrict__ Wh,
    const float* __restrict__ weight, const int* __restrict__ src,
    const int* __restrict__ rev, float* __restrict__ hNext, int nEdges) {
  extern __shared__ _Float16 smem[];            // [256][256] f16
  const int KPAD = HID;
  for (int idx = threadIdx.x; idx < HID * HID; idx += blockDim.x)
    smem[idx] = (_Float16)Wh[idx];
  __syncthreads();

  const int lane = threadIdx.x & 31, wave = threadIdx.x >> 5;
  const int half = lane >> 4, nl = lane & 15;
  const int rowBase = blockIdx.x * 128 + wave * 16;
  int row = rowBase + nl;
  if (row >= nEdges) row = nEdges - 1;
  const int se = src[row];
  const int re = rev[row];
  const float wr = weight[re];
  const float* marow = mAtom + (size_t)se * HID;
  const float* hrow  = hCur + (size_t)re * HID;

  v8f zero = {0.f, 0.f, 0.f, 0.f, 0.f, 0.f, 0.f, 0.f};
  v8f acc[16];
#pragma unroll
  for (int nt = 0; nt < 16; ++nt) acc[nt] = zero;

  for (int kt = 0; kt < KPAD / 32; ++kt) {
    v16h a;
#pragma unroll
    for (int v = 0; v < 8; ++v) {
      int k = kt * 32 + kA_of(v, half);
      a[2 * v]     = (_Float16)(marow[k]     - wr * hrow[k]);
      a[2 * v + 1] = (_Float16)(marow[k + 1] - wr * hrow[k + 1]);
    }
#pragma unroll
    for (int nt = 0; nt < 16; ++nt) {
      const int ncol = nt * 16 + nl;
      v16h b = *(const v16h*)&smem[ncol * KPAD + kt * 32 + half * 16];
      acc[nt] = __builtin_amdgcn_wmma_f32_16x16x32_f16(
          false, a, false, b, (short)0, acc[nt], false, false);
    }
  }
#pragma unroll
  for (int nt = 0; nt < 16; ++nt) {
#pragma unroll
    for (int v = 0; v < 8; ++v) {
      int r = rowBase + v + 8 * half;
      if (r < nEdges) {
        int c = nt * 16 + nl;
        float val = acc[nt][v] + h0[(size_t)r * HID + c];
        hNext[(size_t)r * HID + c] = val > 0.f ? val : 0.f;
      }
    }
  }
}

// ========================= GEMM 3: atom readout =============================
// h_atom[a] = relu( [V[a]; m_final[a]] @ W_o^T + b ), K = 389 padded to 416
__global__ __launch_bounds__(256) void gemm_out_kernel(
    const float* __restrict__ V, const float* __restrict__ mFin,
    const float* __restrict__ Wo, const float* __restrict__ Wb,
    float* __restrict__ hAtom, int nAtoms) {
  extern __shared__ _Float16 smem[];            // [256][416] f16 (208 KB)
  const int KPAD = 416, KREAL = ADIM + HID;     // 389
  for (int idx = threadIdx.x; idx < HID * KPAD; idx += blockDim.x) {
    int n = idx / KPAD, k = idx - n * KPAD;
    smem[idx] = (k < KREAL) ? (_Float16)Wo[n * KREAL + k] : (_Float16)0.0f;
  }
  __syncthreads();

  const int lane = threadIdx.x & 31, wave = threadIdx.x >> 5;
  const int half = lane >> 4, nl = lane & 15;
  const int rowBase = blockIdx.x * 128 + wave * 16;
  int row = rowBase + nl;
  if (row >= nAtoms) row = nAtoms - 1;
  const float* vrow = V + (size_t)row * ADIM;
  const float* mrow = mFin + (size_t)row * HID;

  v8f zero = {0.f, 0.f, 0.f, 0.f, 0.f, 0.f, 0.f, 0.f};
  v8f acc[16];
#pragma unroll
  for (int nt = 0; nt < 16; ++nt) acc[nt] = zero;

  for (int kt = 0; kt < KPAD / 32; ++kt) {
    v16h a;
#pragma unroll
    for (int v = 0; v < 8; ++v) {
      int k = kt * 32 + kA_of(v, half);
      float x0 = (k < ADIM) ? vrow[k] : ((k < KREAL) ? mrow[k - ADIM] : 0.f);
      int k1 = k + 1;
      float x1 = (k1 < ADIM) ? vrow[k1] : ((k1 < KREAL) ? mrow[k1 - ADIM] : 0.f);
      a[2 * v]     = (_Float16)x0;
      a[2 * v + 1] = (_Float16)x1;
    }
#pragma unroll
    for (int nt = 0; nt < 16; ++nt) {
      const int ncol = nt * 16 + nl;
      v16h b = *(const v16h*)&smem[ncol * KPAD + kt * 32 + half * 16];
      acc[nt] = __builtin_amdgcn_wmma_f32_16x16x32_f16(
          false, a, false, b, (short)0, acc[nt], false, false);
    }
  }
#pragma unroll
  for (int nt = 0; nt < 16; ++nt) {
#pragma unroll
    for (int v = 0; v < 8; ++v) {
      int r = rowBase + v + 8 * half;
      if (r < nAtoms) {
        int c = nt * 16 + nl;
        float val = acc[nt][v] + Wb[c];
        hAtom[(size_t)r * HID + c] = val > 0.f ? val : 0.f;
      }
    }
  }
}

// ============================ scatter / utility =============================
__global__ void fill_zero_kernel(float* __restrict__ p, int n) {
  int i = blockIdx.x * blockDim.x + threadIdx.x;
  if (i < n) p[i] = 0.f;
}

// out[dst[e]] += (useW ? w[e] : 1) * h[e]   (float4 per thread, hw fp atomics)
__global__ void scatter_edges_kernel(const float* __restrict__ h,
                                     const float* __restrict__ w,
                                     const int* __restrict__ dst,
                                     float* __restrict__ out, int nEdges,
                                     int useW) {
  int idx = blockIdx.x * blockDim.x + threadIdx.x;
  if (idx >= nEdges * 64) return;
  int e = idx >> 6;
  int c4 = (idx & 63) << 2;
  const float4 v = *(const float4*)(h + (size_t)e * HID + c4);
  float wt = useW ? w[e] : 1.f;
  float* o = out + (size_t)dst[e] * HID + c4;
  unsafeAtomicAdd(o + 0, v.x * wt);
  unsafeAtomicAdd(o + 1, v.y * wt);
  unsafeAtomicAdd(o + 2, v.z * wt);
  unsafeAtomicAdd(o + 3, v.w * wt);
}

// mol[batch[a]] += h_atom[a]
__global__ void scatter_atoms_kernel(const float* __restrict__ hAtom,
                                     const int* __restrict__ batch,
                                     float* __restrict__ mol, int nAtoms) {
  int idx = blockIdx.x * blockDim.x + threadIdx.x;
  if (idx >= nAtoms * 64) return;
  int a = idx >> 6;
  int c4 = (idx & 63) << 2;
  const float4 v = *(const float4*)(hAtom + (size_t)a * HID + c4);
  float* o = mol + (size_t)batch[a] * HID + c4;
  unsafeAtomicAdd(o + 0, v.x);
  unsafeAtomicAdd(o + 1, v.y);
  unsafeAtomicAdd(o + 2, v.z);
  unsafeAtomicAdd(o + 3, v.w);
}

__global__ void copy_batch_kernel(const int* __restrict__ batch,
                                  float* __restrict__ out, int n) {
  int i = blockIdx.x * blockDim.x + threadIdx.x;
  if (i < n) out[i] = (float)batch[i];
}

// ============================== launcher ====================================
extern "C" void kernel_launch(void* const* d_in, const int* in_sizes, int n_in,
                              void* d_out, int out_size, void* d_ws,
                              size_t ws_size, hipStream_t stream) {
  const float* V      = (const float*)d_in[0];
  const float* Eb     = (const float*)d_in[1];
  const float* weight = (const float*)d_in[2];
  const float* Wi     = (const float*)d_in[3];
  const float* Wh     = (const float*)d_in[4];
  const float* Wo     = (const float*)d_in[5];
  const float* Wb     = (const float*)d_in[6];
  const int*   eidx   = (const int*)d_in[7];
  const int*   rev    = (const int*)d_in[8];
  const int*   batch  = (const int*)d_in[9];

  const int NA = in_sizes[0] / ADIM;                    // 200000
  const int NE = in_sizes[1] / BDIM;                    // 400000
  const int G  = (out_size - NA * (HID + 1) - NE * HID) / HID;  // 8000
  const int* src = eidx;
  const int* dst = eidx + NE;

  // output layout: h_atom [NA*256] | atom_batch [NA] | mol_vecs [G*256] | h [NE*256]
  float* out       = (float*)d_out;
  float* out_hatom = out;
  float* out_batch = out + (size_t)NA * HID;
  float* out_mol   = out_batch + NA;
  float* out_h     = out_mol + (size_t)G * HID;

  // workspace: h0 | h_tmp | m_atom
  float* h0     = (float*)d_ws;
  float* h_tmp  = h0 + (size_t)NE * HID;
  float* m_atom = h_tmp + (size_t)NE * HID;

  const dim3 blk(256);
  const int edgeBlocks = (NE + 127) / 128;
  const int atomBlocks = (NA + 127) / 128;
  const int mZeroBlocks = (NA * HID + 255) / 256;
  const int eScatBlocks = (NE * 64 + 255) / 256;

  // h0 = relu([V[src];E] @ Wi^T)
  gemm_h0_kernel<<<edgeBlocks, blk, 160 * HID * 2, stream>>>(V, Eb, Wi, src, h0, NE);

  // 3 message-passing iterations; ping-pong so final h lands in d_out
  const float* hc = h0;
  float* hn[3] = {out_h, h_tmp, out_h};
  for (int t = 0; t < 3; ++t) {
    fill_zero_kernel<<<mZeroBlocks, blk, 0, stream>>>(m_atom, NA * HID);
    scatter_edges_kernel<<<eScatBlocks, blk, 0, stream>>>(hc, weight, dst, m_atom, NE, 1);
    gemm_update_kernel<<<edgeBlocks, blk, HID * HID * 2, stream>>>(
        m_atom, hc, h0, Wh, weight, src, rev, hn[t], NE);
    hc = hn[t];
  }

  // m_final = segment_sum(h, dst), unweighted
  fill_zero_kernel<<<mZeroBlocks, blk, 0, stream>>>(m_atom, NA * HID);
  scatter_edges_kernel<<<eScatBlocks, blk, 0, stream>>>(out_h, weight, dst, m_atom, NE, 0);

  // h_atom = relu([V; m_final] @ Wo^T + b)
  gemm_out_kernel<<<atomBlocks, blk, 416 * HID * 2, stream>>>(
      V, m_atom, Wo, Wb, out_hatom, NA);

  // atom_batch (as float) and mol_vecs pooling
  copy_batch_kernel<<<(NA + 255) / 256, blk, 0, stream>>>(batch, out_batch, NA);
  fill_zero_kernel<<<(G * HID + 255) / 256, blk, 0, stream>>>(out_mol, G * HID);
  scatter_atoms_kernel<<<(NA * 64 + 255) / 256, blk, 0, stream>>>(out_hatom, batch, out_mol, NA);
}
